// LumenLoss_76278619177265
// MI455X (gfx1250) — matrix operand order repaired and checked by reading, not compiled, not run on hardware
//
#include <hip/hip_runtime.h>
#include <hip/hip_bf16.h>

typedef __attribute__((ext_vector_type(16))) __bf16 v16bf;
typedef __attribute__((ext_vector_type(8)))  __bf16 v8bf;
typedef __attribute__((ext_vector_type(8)))  float  v8f;
typedef unsigned int u32x4 __attribute__((ext_vector_type(4)));
typedef int          i32x8 __attribute__((ext_vector_type(8)));
typedef int          i32x4 __attribute__((ext_vector_type(4)));

#define DD 128
#define HH 192
#define WW 192
#define NB (DD*HH*WW)            // 4718592 per batch
#define NALL (2*NB)              // 9437184

// padded bf16 conv-input volumes: halo 3, row stride 208 (covers K=32 over-read)
#define W_S 208
#define H_S 198
#define D_S 134
#define VOL_E (D_S*H_S*W_S)      // 5518656 elements

#define SC 768                   // scalar accumulator base (after 2*384 dice bins)
#define N_ACC (SC + 23)
#define BFRAG_OFF 4096
#define VOLS_OFF  65536
#define ZERO_BYTES ((size_t)VOLS_OFF + 8ull*(unsigned long long)VOL_E*2ull)

__device__ __forceinline__ float wave_sum(float v) {
    #pragma unroll
    for (int m = 16; m > 0; m >>= 1) v += __shfl_xor(v, m, 32);
    return v;
}
__device__ __forceinline__ float sigm(float x) { return 1.f / (1.f + __expf(-x)); }

// ---------------------------------------------------------------------------
// B-fragment table (Toeplitz of CDT filter rows) in contiguous-K WMMA B lane layout,
// built once per call into global ws:   [49 kdkh][32 lane][16 halfs]
// lanes 0-15: N=lane, K=0..15 ; lanes 16-31: N=lane-16, K=16..31
// ---------------------------------------------------------------------------
__global__ void bfrag_init(__bf16* __restrict__ bf) {
    int kk = blockIdx.x, lane = threadIdx.x;          // <<<49, 32>>>
    int kd = kk / 7, kh = kk % 7;
    int n = lane & 15, kbase = (lane >> 4) * 16;
    float dd = (float)((kd - 3) * (kd - 3) + (kh - 3) * (kh - 3));
    __bf16* frag = bf + (kk * 32 + lane) * 16;
    #pragma unroll
    for (int h = 0; h < 16; h++) {
        int t = kbase + h - n;                        // filter tap index kw
        float v = 0.f;
        if (t >= 0 && t < 7)
            v = __expf(-10.f * sqrtf(dd + (float)((t - 3) * (t - 3))));
        frag[h] = (__bf16)v;
    }
}

// ---------------------------------------------------------------------------
// Pass 1: pointwise stats + 4 padded bf16 conv-input volumes
// vols layout: (mode*2 + b)*VOL_E ;  mode 0:yb 1:y_tg 2:1-yb 3:1-y_tg
// ---------------------------------------------------------------------------
__global__ __launch_bounds__(256)
void prep_kernel(const float* __restrict__ x_in, const float* __restrict__ y_out,
                 const float* __restrict__ y_tg, __bf16* __restrict__ vols,
                 float* __restrict__ acc) {
    __shared__ float binsI[384];
    __shared__ float binsC[384];
    for (int t = threadIdx.x; t < 384; t += blockDim.x) { binsI[t] = 0.f; binsC[t] = 0.f; }
    __syncthreads();

    const long long stride = (long long)gridDim.x * blockDim.x;   // 393216 = 2048*192
    const long long i0 = (long long)blockIdx.x * blockDim.x + threadIdx.x;
    float ce = 0.f;
    float st0=0,st1=0, sp0=0,sp1=0, sy0=0,sy1=0, syx0=0,syx1=0, sxx0=0,sxx1=0;
    float li0=0,li1=0, lc0=0,lc1=0;
    for (long long i = i0; i < NALL; i += stride) {
        int ii = (int)i;
        int b = (ii >= NB);
        int r = ii - b * NB;
        int d = r / (HH * WW);
        int r2 = r - d * (HH * WW);
        int h = r2 / WW;
        int w = r2 - h * WW;

        float x = x_in[i], yo = y_out[i], t = y_tg[i];
        float yp = sigm(yo);
        ce += fmaxf(yo, 0.f) - yo * t + log1pf(__expf(-fabsf(yo)));
        float yb = sigm((yp - 0.5f) * 200.f);

        size_t pidx = ((size_t)(d + 3) * H_S + (h + 3)) * W_S + (w + 3);
        size_t vb = (size_t)b * VOL_E;
        vols[0 * 2 * VOL_E + vb + pidx] = (__bf16)yb;
        vols[1 * 2 * VOL_E + vb + pidx] = (__bf16)t;
        vols[2 * 2 * VOL_E + vb + pidx] = (__bf16)(1.f - yb);
        vols[3 * 2 * VOL_E + vb + pidx] = (__bf16)(1.f - t);

        if (b == 0) { st0+=t; sp0+=yp; sy0+=yb; syx0+=yb*x; sxx0+=yb*x*x; li0+=yp*t; lc0+=yp+t; }
        else        { st1+=t; sp1+=yp; sy1+=yb; syx1+=yb*x; sxx1+=yb*x*x; li1+=yp*t; lc1+=yp+t; }
    }
    int w = (int)(i0 % WW);                 // stride % 192 == 0 -> fixed per thread
    atomicAdd(&binsI[w], li0);       atomicAdd(&binsI[192 + w], li1);
    atomicAdd(&binsC[w], lc0);       atomicAdd(&binsC[192 + w], lc1);

    float rr[11] = {ce, st0, st1, sp0, sp1, sy0, sy1, syx0, syx1, sxx0, sxx1};
    #pragma unroll
    for (int k = 0; k < 11; k++) rr[k] = wave_sum(rr[k]);
    if ((threadIdx.x & 31) == 0)
        #pragma unroll
        for (int k = 0; k < 11; k++) atomicAdd(&acc[SC + k], rr[k]);

    __syncthreads();
    for (int t = threadIdx.x; t < 384; t += blockDim.x) {
        atomicAdd(&acc[t],       binsI[t]);
        atomicAdd(&acc[384 + t], binsC[t]);
    }
}

// ---------------------------------------------------------------------------
// Pass 2: 7x7x7 CDT conv via bf16 WMMA; TDM async-tensor loads feed LDS
// grid = (12, 12, 4 modes * 2 batches * 32 d-tiles), 128 threads (4 waves)
// ---------------------------------------------------------------------------
#define TP 10
#define TR 22
#define TC 32

__global__ __launch_bounds__(128)
void cdt_conv_kernel(const __bf16* __restrict__ vols, const __bf16* __restrict__ bfrag_g,
                     const float* __restrict__ y_tg, const float* __restrict__ y_out,
                     const float* __restrict__ x_in, float* __restrict__ acc) {
    __shared__ __attribute__((aligned(128))) __bf16 tileS[TP * TR * TC];    // 14080 B
    __shared__ __attribute__((aligned(128))) __bf16 bfragS[49 * 32 * 16];   // 50176 B

    const int zb   = blockIdx.z;
    const int mode = zb >> 6;
    const int rem  = zb & 63;
    const int b    = rem >> 5;
    const int dt   = rem & 31;
    const int w0 = blockIdx.x * 16, h0 = blockIdx.y * 16, d0 = dt * 4;
    const int tid = threadIdx.x;
    const long long base_b = (long long)b * NB;

    if (tid < 32) {
        i32x8 gz = {0, 0, 0, 0, 0, 0, 0, 0};
        // ---- D# for input tile: 3D 32x22x10 bf16 from padded volume ----
        const __bf16* src = vols + (size_t)(mode * 2 + b) * VOL_E;
        unsigned long long ga =
            (unsigned long long)(size_t)src +
            2ull * (((unsigned long long)d0 * H_S + h0) * W_S + w0);
        u32x4 g0; i32x8 g1; i32x4 g2, g3;
        g0[0] = 1u;                                        // count=1
        g0[1] = (unsigned)(size_t)&tileS[0];               // lds_addr (bytes)
        g0[2] = (unsigned)(ga & 0xffffffffull);
        g0[3] = ((unsigned)(ga >> 32) & 0x01ffffffu) | 0x80000000u;  // type=2
        g1[0] = (int)(1u << 16);                           // data_size=1 (2B)
        g1[1] = (int)((unsigned)W_S << 16);                // tensor_dim0 lo
        g1[2] = (int)((unsigned)H_S << 16);                // dim0 hi | tensor_dim1 lo
        g1[3] = (int)(32u << 16);                          // dim1 hi | tile_dim0=32
        g1[4] = (int)(22u | (10u << 16));                  // tile_dim1=22 | tile_dim2=10
        g1[5] = W_S;                                       // tensor_dim0_stride
        g1[6] = (int)(((unsigned)(H_S * W_S) & 0xffffu) << 16);  // | dim1_stride lo
        g1[7] = (int)((unsigned)(H_S * W_S) >> 16);        // dim1_stride hi
        g2[0] = D_S;                                       // tensor_dim2
        g2[1] = 0;
        g2[2] = D_S * H_S * W_S;                           // tensor_dim2_stride
        g2[3] = 0;
        g3[0] = 0; g3[1] = 0; g3[2] = 0; g3[3] = 0;
        __builtin_amdgcn_tensor_load_to_lds(g0, g1, g2, g3, gz, 0);

        // ---- D# for B-fragment table: 1D copy, 6272 x 8B ----
        unsigned long long gb = (unsigned long long)(size_t)bfrag_g;
        u32x4 h0v; i32x8 h1v; i32x4 h2v, h3v;
        h0v[0] = 1u;
        h0v[1] = (unsigned)(size_t)&bfragS[0];
        h0v[2] = (unsigned)(gb & 0xffffffffull);
        h0v[3] = ((unsigned)(gb >> 32) & 0x01ffffffu) | 0x80000000u;
        h1v[0] = (int)(3u << 16);                          // data_size=3 (8B)
        h1v[1] = (int)(6272u << 16);                       // tensor_dim0=6272 lo
        h1v[2] = (int)(1u << 16);                          // tensor_dim1=1
        h1v[3] = (int)(6272u << 16);                       // tile_dim0=6272
        h1v[4] = 0;                                        // tile_dim1=0 (1D)
        h1v[5] = 6272;                                     // stride0
        h1v[6] = (int)((6272u & 0xffffu) << 16);           // stride1 lo
        h1v[7] = 0;
        h2v[0] = 1; h2v[1] = 0; h2v[2] = 0; h2v[3] = 0;
        h3v[0] = 0; h3v[1] = 0; h3v[2] = 0; h3v[3] = 0;
        __builtin_amdgcn_tensor_load_to_lds(h0v, h1v, h2v, h3v, gz, 0);

        __builtin_amdgcn_s_wait_tensorcnt(0);
    }
    __syncthreads();

    const int wave = tid >> 5, lane = tid & 31;
    const int d = d0 + wave;
    const int mn = lane & 15, sel = lane >> 4;

    // A contiguous-K layout (matches B table): lane holds row M=lane&15; sel 0 -> K{0..15}, sel 1 -> K{16..31}
    const __bf16* paBase = tileS + ((size_t)wave * TR + mn) * TC + sel * 16;
    const __bf16* pbBase = bfragS + (size_t)lane * 16;

    v8f cacc = {};
    #pragma unroll
    for (int kd = 0; kd < 7; kd++) {
        const __bf16* pa = paBase + kd * (TR * TC);
        const __bf16* pb = pbBase + kd * 7 * (32 * 16);
        #pragma unroll
        for (int kh = 0; kh < 7; kh++) {
            v8bf alo = *(const v8bf*)(pa + kh * TC);
            v8bf ahi = *(const v8bf*)(pa + kh * TC + 8);
            v8bf blo = *(const v8bf*)(pb + kh * (32 * 16));
            v8bf bhi = *(const v8bf*)(pb + kh * (32 * 16) + 8);
            v16bf A, B;
            #pragma unroll
            for (int q = 0; q < 8; q++) { A[q]=alo[q]; A[q+8]=ahi[q]; B[q]=blo[q]; B[q+8]=bhi[q]; }
            cacc = __builtin_amdgcn_wmma_f32_16x16x32_bf16(false, A, false, B, (short)0, cacc,
                                                           false, false);
        }
    }

    // Fused epilogue: z = -gamma*log(conv+eps) + mode-specific reductions
    float lm = 0.f, ly4 = 0.f, ly4x = 0.f, ly4xx = 0.f;
    #pragma unroll
    for (int j = 0; j < 8; j++) {
        int h = h0 + j + sel * 8, w = w0 + mn;            // C/D: VGPR j -> M=j (+8 hi lanes)
        long long idx = base_b + ((long long)d * HH + h) * WW + w;
        float z = -0.1f * __logf(cacc[j] + 1e-6f);
        if (mode == 0) {
            lm += y_tg[idx] * z;
            float y3 = fmaxf(3.f - z, 0.f);
            float g  = sigm(y3 * 200.f);
            float y4 = sigm(((3.f - y3) * g - 0.1f) * 200.f);
            float x  = x_in[idx];
            ly4 += y4; ly4x += y4 * x; ly4xx += y4 * x * x;
        } else if (mode == 1) {
            lm += sigm(y_out[idx]) * z;
        } else if (mode == 2) {
            lm += (1.f - y_tg[idx]) * z;
        } else {
            lm += (1.f - sigm(y_out[idx])) * z;
        }
    }
    lm = wave_sum(lm);
    if (mode == 0) { ly4 = wave_sum(ly4); ly4x = wave_sum(ly4x); ly4xx = wave_sum(ly4xx); }
    if (lane == 0) {
        atomicAdd(&acc[SC + 17 + mode], lm);
        if (mode == 0) {
            atomicAdd(&acc[SC + 11 + b], ly4);
            atomicAdd(&acc[SC + 13 + b], ly4x);
            atomicAdd(&acc[SC + 15 + b], ly4xx);
        }
    }
}

// ---------------------------------------------------------------------------
// Pass 3: 3x3x3 Laplacian + Scharr magnitude on padded yb (no bounds checks)
// ---------------------------------------------------------------------------
__global__ __launch_bounds__(256)
void edge_kernel(const __bf16* __restrict__ vols, float* __restrict__ acc) {
    const long long stride = (long long)gridDim.x * blockDim.x;
    const float hs[3] = {47.f / 256.f, 162.f / 256.f, 47.f / 256.f};
    const float pr[3] = {1.f, 0.f, -1.f};
    float s12 = 0.f, s1 = 0.f;
    for (long long i = (long long)blockIdx.x * blockDim.x + threadIdx.x; i < NALL; i += stride) {
        int ii = (int)i;
        int bb = (ii >= NB);
        int r = ii - bb * NB;
        int d = r / (HH * WW);
        int r2 = r - d * (HH * WW);
        int h = r2 / WW;
        int w = r2 - h * WW;
        const __bf16* yb = vols + (size_t)bb * VOL_E;     // mode 0 volume
        float lap = 0.f, g1 = 0.f, g2 = 0.f, g3 = 0.f;
        #pragma unroll
        for (int kd = 0; kd < 3; kd++)
        #pragma unroll
        for (int kh = 0; kh < 3; kh++)
        #pragma unroll
        for (int kw = 0; kw < 3; kw++) {
            float v = (float)yb[((size_t)(d + 2 + kd) * H_S + (h + 2 + kh)) * W_S + (w + 2 + kw)];
            int na = (kd != 1) + (kh != 1) + (kw != 1);
            float lw = (na == 3 ? 2.f : na == 2 ? 3.f : na == 1 ? 6.f : -88.f) * (1.f / 26.f);
            lap += lw * v;
            g1 += hs[kd] * hs[kh] * pr[kw] * v;
            g2 += hs[kd] * pr[kh] * hs[kw] * v;
            g3 += pr[kd] * hs[kh] * hs[kw] * v;
        }
        float y1 = sigm((lap - 0.5f) * 200.f);
        float y2 = sqrtf(g1 * g1 + g2 * g2 + g3 * g3 + 1e-6f);
        s12 += y1 * y2; s1 += y1;
    }
    s12 = wave_sum(s12); s1 = wave_sum(s1);
    if ((threadIdx.x & 31) == 0) { atomicAdd(&acc[SC + 21], s12); atomicAdd(&acc[SC + 22], s1); }
}

// ---------------------------------------------------------------------------
// Pass 4: combine accumulators into final scalar loss
// ---------------------------------------------------------------------------
__global__ void finalize_kernel(const float* __restrict__ acc,
                                const float* __restrict__ gamma_ace,
                                float* __restrict__ out) {
    if (threadIdx.x != 0 || blockIdx.x != 0) return;
    const float Nf = (float)NALL, Nb = (float)NB;
    float dice = 0.f;
    for (int i = 0; i < 384; i++)
        dice += 1.f - 2.f * acc[i] / (acc[384 + i] + 1e-6f);
    dice /= 384.f;
    const float* s = acc + SC;
    float loss_ce = s[0] / Nf;
    float st = s[1] + s[2], sp = s[3] + s[4];
    float dl1 = (s[17] / Nf) / (st / Nf + 1e-6f);
    float dl2 = (s[18] / Nf) / (sp / Nf + 1e-6f);
    float dl3 = (s[19] / Nf) / ((Nf - st) / Nf + 1e-6f);
    float dl4 = (s[20] / Nf) / ((Nf - sp) / Nf + 1e-6f);
    float loss_dt = dl1 + dl2 + dl3 + dl4;
    float E1n = 0.f, E2n = 0.f;
    for (int b = 0; b < 2; b++) {
        float sy = s[5 + b], syx = s[7 + b], sxx = s[9 + b];
        float l1 = (syx / Nb) / (sy / Nb + 1e-6f);
        E1n += sxx - 2.f * l1 * syx + l1 * l1 * sy;
        float sy4 = s[11 + b], sy4x = s[13 + b], sy4xx = s[15 + b];
        float l2 = (sy4x / Nb) / (sy4 / Nb + 1e-6f);
        E2n += sy4xx - 2.f * l2 * sy4x + l2 * l2 * sy4;
    }
    float E1 = (E1n / Nf) / ((s[5] + s[6]) / Nf + 1e-6f);
    float E2 = (E2n / Nf) / ((s[11] + s[12]) / Nf + 1e-6f);
    float S  = (s[21] / Nf) / (s[22] / Nf + 1e-6f);
    float ace = E1 + E2 + gamma_ace[0] * S;
    out[0] = loss_ce + dice + 0.1f * loss_dt + 0.1f * ace;
}

// ---------------------------------------------------------------------------
extern "C" void kernel_launch(void* const* d_in, const int* in_sizes, int n_in,
                              void* d_out, int out_size, void* d_ws, size_t ws_size,
                              hipStream_t stream) {
    const float* x_in  = (const float*)d_in[0];
    const float* y_out = (const float*)d_in[1];
    const float* y_tg  = (const float*)d_in[2];
    const float* gamma = (const float*)d_in[4];

    float*  acc     = (float*)d_ws;
    __bf16* bfrag_g = (__bf16*)((char*)d_ws + BFRAG_OFF);
    __bf16* vols    = (__bf16*)((char*)d_ws + VOLS_OFF);

    (void)hipMemsetAsync(d_ws, 0, ZERO_BYTES, stream);
    bfrag_init<<<49, 32, 0, stream>>>(bfrag_g);
    prep_kernel<<<1536, 256, 0, stream>>>(x_in, y_out, y_tg, vols, acc);
    cdt_conv_kernel<<<dim3(12, 12, 256), 128, 0, stream>>>(vols, bfrag_g, y_tg, y_out, x_in, acc);
    edge_kernel<<<1536, 256, 0, stream>>>(vols, acc);
    finalize_kernel<<<1, 32, 0, stream>>>(acc, gamma, (float*)d_out);
}
